// BatchedViT_50130858279815
// MI455X (gfx1250) — compile-verified, matrix operand order checked by reading, not run on hardware
//
#include <hip/hip_runtime.h>
#include <hip/hip_bf16.h>

// ---------------------------------------------------------------------------
// ViT-Base forward for MI455X (gfx1250).
// All GEMMs: v_wmma_f32_16x16x32_f16, fp32 accum, 256x128 block tile,
// 8 waves x (64x64) register tiles = 16 WMMA per K-step per wave.
// Fast-path staging: float4 global loads -> packed f16 -> ds_store_b64.
// ---------------------------------------------------------------------------

typedef _Float16 v2h  __attribute__((ext_vector_type(2)));
typedef _Float16 v4h  __attribute__((ext_vector_type(4)));
typedef _Float16 v8h  __attribute__((ext_vector_type(8)));
typedef _Float16 v16h __attribute__((ext_vector_type(16)));
typedef float    v8f  __attribute__((ext_vector_type(8)));

#define BM 256
#define BN 128
#define KT 32
#define APITCH 40   // halves; 80B row pitch -> [m][0/8/16/24] all 16B aligned
#define BPITCH 40

// model dims
#define B_    32
#define C_    3
#define HW_   224
#define P_    16
#define D_    768
#define L_    12
#define HEADS 12
#define DH_   64
#define MLP_  3072
#define NC_   1000
#define NTOK  197
#define NPATCH 196
#define ROWS  (B_ * NTOK)        // 6304
#define PROWS (B_ * NPATCH)      // 6272

struct GemmP {
    const float* A; const float* Bm; const float* bias; const float* Res; float* C;
    int M, N, K;
    int lda, ldb, ldc;
    long long sAo, sAi, sBo, sBi, sCo, sCi;
    int inner;
    int transB;
    float alpha;
    int act;        // 0 = none, 1 = tanh-GELU
};

__device__ __forceinline__ float gelu_f(float x) {
    const float c = 0.7978845608028654f;  // sqrt(2/pi)
    float t = tanhf(c * (x + 0.044715f * x * x * x));
    return 0.5f * x * (1.0f + t);
}

__global__ __launch_bounds__(256)
void vit_gemm_wmma(GemmP p) {
    __shared__ __align__(16) _Float16 At[BM * APITCH];   // 20 KB
    __shared__ __align__(16) _Float16 Bt[BN * BPITCH];   // 10 KB  (stored [n][k])

    const int tid  = threadIdx.x;
    const int lane = tid & 31;
    const int wave = tid >> 5;       // 0..7
    const int wm   = wave >> 1;      // 0..3  -> rows wm*64
    const int wn   = wave & 1;       // 0..1  -> cols wn*64

    const int m0 = blockIdx.y * BM;
    const int n0 = blockIdx.x * BN;

    const long long zo = (long long)blockIdx.z / p.inner;
    const long long zi = (long long)blockIdx.z % p.inner;
    const float* Ab = p.A  + zo * p.sAo + zi * p.sAi;
    const float* Bb = p.Bm + zo * p.sBo + zi * p.sBi;
    float*       Cb = p.C  + zo * p.sCo + zi * p.sCi;
    const float* Rb = p.Res ? (p.Res + zo * p.sCo + zi * p.sCi) : nullptr;

    // uniform fast-path conditions
    const bool avec = ((p.lda & 3) == 0) && ((((unsigned long long)Ab) & 15ull) == 0);
    const bool bvec = ((p.ldb & 3) == 0) && ((((unsigned long long)Bb) & 15ull) == 0);
    const bool mfull = (m0 + BM) <= p.M;
    const bool nfull = (n0 + BN) <= p.N;

    v8f acc[4][4];
#pragma unroll
    for (int i = 0; i < 4; ++i)
#pragma unroll
        for (int j = 0; j < 4; ++j)
#pragma unroll
            for (int r = 0; r < 8; ++r) acc[i][j][r] = 0.0f;

    const int kTiles = (p.K + KT - 1) / KT;
    for (int kt = 0; kt < kTiles; ++kt) {
        const int k0 = kt * KT;
        const bool kfull = (k0 + KT) <= p.K;
        __syncthreads();

        // ---------------- stage A tile: 256 rows x 32 k, one row per thread --
        {
            const int r  = tid;
            const int gm = m0 + r;
            const float* src = Ab + (long long)gm * p.lda + k0;
            _Float16* dst = &At[r * APITCH];
            if ((mfull || gm < p.M) && kfull) {
                if (avec) {
#pragma unroll
                    for (int u = 0; u < 8; ++u) {
                        float4 f = reinterpret_cast<const float4*>(src)[u];
                        v4h h; h[0] = (_Float16)f.x; h[1] = (_Float16)f.y;
                               h[2] = (_Float16)f.z; h[3] = (_Float16)f.w;
                        *reinterpret_cast<v4h*>(&dst[u * 4]) = h;
                    }
                } else {
#pragma unroll
                    for (int u = 0; u < 32; ++u) dst[u] = (_Float16)src[u];
                }
            } else {
#pragma unroll
                for (int u = 0; u < 32; ++u) {
                    float v = ((gm < p.M) && (k0 + u) < p.K) ? src[u] : 0.0f;
                    dst[u] = (_Float16)v;
                }
            }
        }
        // ---------------- stage B tile transposed -> Bt[n][k] ----------------
        if (p.transB) {
            // Bsrc is [N,K] row major: Bt[n][k] = Bsrc[n*ldb + k]; contiguous in k
            const int n  = tid >> 1;             // 0..127
            const int kb = (tid & 1) * 16;       // 0,16
            const int gn = n0 + n;
            const float* src = Bb + (long long)gn * p.ldb + k0 + kb;
            _Float16* dst = &Bt[n * BPITCH + kb];
            if ((nfull || gn < p.N) && kfull) {
                if (bvec) {
#pragma unroll
                    for (int u = 0; u < 4; ++u) {
                        float4 f = reinterpret_cast<const float4*>(src)[u];
                        v4h h; h[0] = (_Float16)f.x; h[1] = (_Float16)f.y;
                               h[2] = (_Float16)f.z; h[3] = (_Float16)f.w;
                        *reinterpret_cast<v4h*>(&dst[u * 4]) = h;
                    }
                } else {
#pragma unroll
                    for (int u = 0; u < 16; ++u) dst[u] = (_Float16)src[u];
                }
            } else {
#pragma unroll
                for (int u = 0; u < 16; ++u) {
                    float v = ((gn < p.N) && (k0 + kb + u) < p.K) ? src[u] : 0.0f;
                    dst[u] = (_Float16)v;
                }
            }
        } else {
            // Bsrc is [K,N] row major: read two k-rows, pack (k,k+1) per b32 store
            const int kp = tid >> 4;             // 0..15 -> k = 2*kp, 2*kp+1
            const int nb = (tid & 15) * 8;       // 0..120
            const int gk0 = k0 + 2 * kp, gk1 = gk0 + 1;
            const float* s0 = Bb + (long long)gk0 * p.ldb + n0 + nb;
            const float* s1 = Bb + (long long)gk1 * p.ldb + n0 + nb;
            if (nfull && kfull) {
                if (bvec) {
                    float4 a0 = reinterpret_cast<const float4*>(s0)[0];
                    float4 a1 = reinterpret_cast<const float4*>(s0)[1];
                    float4 c0 = reinterpret_cast<const float4*>(s1)[0];
                    float4 c1 = reinterpret_cast<const float4*>(s1)[1];
                    float r0[8] = {a0.x,a0.y,a0.z,a0.w,a1.x,a1.y,a1.z,a1.w};
                    float r1[8] = {c0.x,c0.y,c0.z,c0.w,c1.x,c1.y,c1.z,c1.w};
#pragma unroll
                    for (int u = 0; u < 8; ++u) {
                        v2h h; h[0] = (_Float16)r0[u]; h[1] = (_Float16)r1[u];
                        *reinterpret_cast<v2h*>(&Bt[(nb + u) * BPITCH + 2 * kp]) = h;
                    }
                } else {
#pragma unroll
                    for (int u = 0; u < 8; ++u) {
                        v2h h; h[0] = (_Float16)s0[u]; h[1] = (_Float16)s1[u];
                        *reinterpret_cast<v2h*>(&Bt[(nb + u) * BPITCH + 2 * kp]) = h;
                    }
                }
            } else {
#pragma unroll
                for (int u = 0; u < 8; ++u) {
                    int gn = n0 + nb + u;
                    float v0 = ((gn < p.N) && gk0 < p.K) ? s0[u] : 0.0f;
                    float v1 = ((gn < p.N) && gk1 < p.K) ? s1[u] : 0.0f;
                    v2h h; h[0] = (_Float16)v0; h[1] = (_Float16)v1;
                    *reinterpret_cast<v2h*>(&Bt[(nb + u) * BPITCH + 2 * kp]) = h;
                }
            }
        }
        __syncthreads();

        // prefetch next K tiles (global_prefetch_b8)
        if (kt + 1 < kTiles) {
            int gm = m0 + tid;
            if (gm < p.M)
                __builtin_prefetch(Ab + (long long)gm * p.lda + (k0 + KT), 0, 0);
            int gn = n0 + (tid >> 1);
            if (gn < p.N)
                __builtin_prefetch(p.transB ? (Bb + (long long)gn * p.ldb + k0 + KT)
                                            : (Bb + (long long)(k0 + KT) * p.ldb + n0),
                                   0, 0);
        }

        // ---------------- fragments (16-bit A/B WMMA VGPR layout) -------------
        const int off = (lane >= 16) ? 8 : 0;     // lanes 16..31 hold K+8
        const int mr  = lane & 15;
        v16h afr[4], bfr[4];
#pragma unroll
        for (int i = 0; i < 4; ++i) {
            int m = wm * 64 + i * 16 + mr;
            v8h lo = *reinterpret_cast<const v8h*>(&At[m * APITCH + off]);
            v8h hi = *reinterpret_cast<const v8h*>(&At[m * APITCH + off + 16]);
            afr[i] = __builtin_shufflevector(lo, hi, 0,1,2,3,4,5,6,7,8,9,10,11,12,13,14,15);
        }
#pragma unroll
        for (int j = 0; j < 4; ++j) {
            int n = wn * 64 + j * 16 + mr;
            v8h lo = *reinterpret_cast<const v8h*>(&Bt[n * BPITCH + off]);
            v8h hi = *reinterpret_cast<const v8h*>(&Bt[n * BPITCH + off + 16]);
            bfr[j] = __builtin_shufflevector(lo, hi, 0,1,2,3,4,5,6,7,8,9,10,11,12,13,14,15);
        }
#pragma unroll
        for (int i = 0; i < 4; ++i)
#pragma unroll
            for (int j = 0; j < 4; ++j)
                acc[i][j] = __builtin_amdgcn_wmma_f32_16x16x32_f16(
                    false, afr[i], false, bfr[j], (short)0, acc[i][j], false, false);
    }

    // ---------------- epilogue ----------------
    const int hi8 = (lane >= 16) ? 8 : 0;
#pragma unroll
    for (int i = 0; i < 4; ++i) {
#pragma unroll
        for (int j = 0; j < 4; ++j) {
#pragma unroll
            for (int r = 0; r < 8; ++r) {
                int row = m0 + wm * 64 + i * 16 + r + hi8;
                int col = n0 + wn * 64 + j * 16 + (lane & 15);
                if (row < p.M && col < p.N) {
                    float v = acc[i][j][r] * p.alpha;
                    if (p.bias) v += p.bias[col];
                    if (p.act == 1) v = gelu_f(v);
                    long long ci = (long long)row * p.ldc + col;
                    if (Rb) v += Rb[ci];
                    Cb[ci] = v;
                }
            }
        }
    }
}

// ---------------- support kernels ----------------

__global__ __launch_bounds__(256)
void vit_im2col(const float* __restrict__ imgs, float* __restrict__ X) {
    long long idx = (long long)blockIdx.x * blockDim.x + threadIdx.x;
    long long total = (long long)PROWS * D_;
    if (idx >= total) return;
    int col = (int)(idx % D_);
    int row = (int)(idx / D_);
    int b  = row / NPATCH;
    int pp = row % NPATCH;
    int pi = pp / 14, pj = pp % 14;
    int c  = col / (P_ * P_);
    int rem = col % (P_ * P_);
    int ph = rem / P_, pw = rem % P_;
    X[idx] = imgs[(((long long)b * C_ + c) * HW_ + (pi * P_ + ph)) * HW_ + (pj * P_ + pw)];
}

__global__ __launch_bounds__(256)
void vit_assemble(const float* __restrict__ patches, const float* __restrict__ cls,
                  const float* __restrict__ pos, float* __restrict__ x) {
    long long idx = (long long)blockIdx.x * blockDim.x + threadIdx.x;
    long long total = (long long)ROWS * D_;
    if (idx >= total) return;
    int d = (int)(idx % D_);
    int t = (int)((idx / D_) % NTOK);
    int b = (int)(idx / ((long long)D_ * NTOK));
    float v = (t == 0) ? cls[d] : patches[((long long)b * NPATCH + (t - 1)) * D_ + d];
    x[idx] = v + pos[(long long)t * D_ + d];
}

__global__ __launch_bounds__(256)
void vit_layernorm(const float* __restrict__ in, const float* __restrict__ g,
                   const float* __restrict__ b, float* __restrict__ out, int D) {
    __shared__ float red[256];
    long long row = blockIdx.x;
    const float* p = in + row * (long long)D;
    float* q = out + row * (long long)D;
    int t = threadIdx.x;
    float s = 0.0f;
    for (int d = t; d < D; d += 256) s += p[d];
    red[t] = s; __syncthreads();
    for (int sh = 128; sh > 0; sh >>= 1) { if (t < sh) red[t] += red[t + sh]; __syncthreads(); }
    float mu = red[0] / D; __syncthreads();
    float s2 = 0.0f;
    for (int d = t; d < D; d += 256) { float z = p[d] - mu; s2 += z * z; }
    red[t] = s2; __syncthreads();
    for (int sh = 128; sh > 0; sh >>= 1) { if (t < sh) red[t] += red[t + sh]; __syncthreads(); }
    float rstd = rsqrtf(red[0] / D + 1e-5f);
    for (int d = t; d < D; d += 256) q[d] = (p[d] - mu) * rstd * g[d] + b[d];
}

__global__ __launch_bounds__(256)
void vit_softmax(float* __restrict__ x, int cols) {
    __shared__ float red[256];
    float* p = x + (long long)blockIdx.x * cols;
    int t = threadIdx.x;
    float v = (t < cols) ? p[t] : -3.4e38f;
    red[t] = v; __syncthreads();
    for (int s = 128; s > 0; s >>= 1) { if (t < s) red[t] = fmaxf(red[t], red[t + s]); __syncthreads(); }
    float mx = red[0]; __syncthreads();
    float e = (t < cols) ? __expf(v - mx) : 0.0f;
    red[t] = e; __syncthreads();
    for (int s = 128; s > 0; s >>= 1) { if (t < s) red[t] += red[t + s]; __syncthreads(); }
    float sum = red[0];
    if (t < cols) p[t] = e / sum;
}

__global__ __launch_bounds__(256)
void vit_meanpool(const float* __restrict__ x, float* __restrict__ out) {
    int idx = blockIdx.x * blockDim.x + threadIdx.x;
    if (idx >= B_ * D_) return;
    int b = idx / D_, d = idx % D_;
    float s = 0.0f;
    for (int t = 0; t < NTOK; ++t)
        s += x[((long long)b * NTOK + t) * D_ + d];
    out[idx] = s * (1.0f / NTOK);
}

// ---------------- host orchestration ----------------

static void launch_gemm(const float* A, int lda, const float* Bm, int ldb, int transB,
                        const float* bias, const float* Res, float* C, int ldc,
                        int M, int N, int K, float alpha, int act,
                        long long sAo, long long sAi, long long sBo, long long sBi,
                        long long sCo, long long sCi, int outer, int inner,
                        hipStream_t stream) {
    GemmP p;
    p.A = A; p.Bm = Bm; p.bias = bias; p.Res = Res; p.C = C;
    p.M = M; p.N = N; p.K = K; p.lda = lda; p.ldb = ldb; p.ldc = ldc;
    p.sAo = sAo; p.sAi = sAi; p.sBo = sBo; p.sBi = sBi; p.sCo = sCo; p.sCi = sCi;
    p.inner = inner; p.transB = transB; p.alpha = alpha; p.act = act;
    dim3 grid((N + BN - 1) / BN, (M + BM - 1) / BM, outer * inner);
    vit_gemm_wmma<<<grid, dim3(256), 0, stream>>>(p);
}

extern "C" void kernel_launch(void* const* d_in, const int* in_sizes, int n_in,
                              void* d_out, int out_size, void* d_ws, size_t ws_size,
                              hipStream_t stream) {
    (void)in_sizes; (void)n_in; (void)out_size; (void)ws_size;
    const float* imgs    = (const float*)d_in[0];
    const float* patch_w = (const float*)d_in[1];
    const float* patch_b = (const float*)d_in[2];
    const float* cls_tok = (const float*)d_in[3];
    const float* pos_emb = (const float*)d_in[4];
    const float* ln1_g   = (const float*)d_in[5];
    const float* ln1_b   = (const float*)d_in[6];
    const float* wq      = (const float*)d_in[7];
    const float* bq      = (const float*)d_in[8];
    const float* wk      = (const float*)d_in[9];
    const float* bk      = (const float*)d_in[10];
    const float* wv      = (const float*)d_in[11];
    const float* bv      = (const float*)d_in[12];
    const float* wo      = (const float*)d_in[13];
    const float* bo      = (const float*)d_in[14];
    const float* ln2_g   = (const float*)d_in[15];
    const float* ln2_b   = (const float*)d_in[16];
    const float* w1      = (const float*)d_in[17];
    const float* b1      = (const float*)d_in[18];
    const float* w2      = (const float*)d_in[19];
    const float* b2      = (const float*)d_in[20];
    const float* hn_g    = (const float*)d_in[21];
    const float* hn_b    = (const float*)d_in[22];
    const float* hw      = (const float*)d_in[23];
    const float* hb      = (const float*)d_in[24];
    float* out = (float*)d_out;

    char* w = (char*)d_ws;
    auto alloc = [&](size_t bytes) { char* p = w; w += (bytes + 255) & ~(size_t)255; return p; };
    const size_t SZ_ACT = (size_t)ROWS * D_ * 4;
    const size_t SZ_BIG = (size_t)ROWS * MLP_ * 4;
    float* big   = (float*)alloc(SZ_BIG);   // im2col X / attn scores / MLP hidden
    float* xbuf  = (float*)alloc(SZ_ACT);
    float* hbuf  = (float*)alloc(SZ_ACT);
    float* qbuf  = (float*)alloc(SZ_ACT);
    float* kbuf  = (float*)alloc(SZ_ACT);
    float* vbuf  = (float*)alloc(SZ_ACT);
    float* obuf  = (float*)alloc(SZ_ACT);
    float* pool  = (float*)alloc((size_t)B_ * D_ * 4);
    float* pool2 = (float*)alloc((size_t)B_ * D_ * 4);

    const long long sTok = (long long)NTOK * D_;
    const long long sSco = (long long)HEADS * NTOK * NTOK;
    const long long sScI = (long long)NTOK * NTOK;
    const float scale = 0.125f;   // DH^-0.5

    // 1) patch embedding: im2col + (X @ patch_w^T + b) -> hbuf, assemble tokens
    {
        long long tot = (long long)PROWS * D_;
        vit_im2col<<<dim3((unsigned)((tot + 255) / 256)), dim3(256), 0, stream>>>(imgs, big);
        launch_gemm(big, D_, patch_w, D_, /*transB=*/1, patch_b, nullptr, hbuf, D_,
                    PROWS, D_, D_, 1.0f, 0, 0,0, 0,0, 0,0, 1, 1, stream);
        long long tot2 = (long long)ROWS * D_;
        vit_assemble<<<dim3((unsigned)((tot2 + 255) / 256)), dim3(256), 0, stream>>>(hbuf, cls_tok, pos_emb, xbuf);
    }

    // 2) transformer layers
    for (int i = 0; i < L_; ++i) {
        const float* Wq = wq + (long long)i * D_ * D_;
        const float* Wk = wk + (long long)i * D_ * D_;
        const float* Wv = wv + (long long)i * D_ * D_;
        const float* Wo = wo + (long long)i * D_ * D_;
        const float* W1 = w1 + (long long)i * D_ * MLP_;
        const float* W2 = w2 + (long long)i * MLP_ * D_;

        vit_layernorm<<<dim3(ROWS), dim3(256), 0, stream>>>(xbuf, ln1_g + i * D_, ln1_b + i * D_, hbuf, D_);

        launch_gemm(hbuf, D_, Wq, D_, 0, bq + i * D_, nullptr, qbuf, D_,
                    ROWS, D_, D_, 1.0f, 0, 0,0, 0,0, 0,0, 1, 1, stream);
        launch_gemm(hbuf, D_, Wk, D_, 0, bk + i * D_, nullptr, kbuf, D_,
                    ROWS, D_, D_, 1.0f, 0, 0,0, 0,0, 0,0, 1, 1, stream);
        launch_gemm(hbuf, D_, Wv, D_, 0, bv + i * D_, nullptr, vbuf, D_,
                    ROWS, D_, D_, 1.0f, 0, 0,0, 0,0, 0,0, 1, 1, stream);

        // scores = scale * q @ k^T, batched over (b, head)
        launch_gemm(qbuf, D_, kbuf, D_, /*transB=*/1, nullptr, nullptr, big, NTOK,
                    NTOK, NTOK, DH_, scale, 0,
                    sTok, DH_, sTok, DH_, sSco, sScI, B_, HEADS, stream);
        vit_softmax<<<dim3(B_ * HEADS * NTOK), dim3(256), 0, stream>>>(big, NTOK);

        // o = att @ v
        launch_gemm(big, NTOK, vbuf, D_, 0, nullptr, nullptr, obuf, D_,
                    NTOK, DH_, NTOK, 1.0f, 0,
                    sSco, sScI, sTok, DH_, sTok, DH_, B_, HEADS, stream);

        // x = x + o @ wo + bo
        launch_gemm(obuf, D_, Wo, D_, 0, bo + i * D_, xbuf, xbuf, D_,
                    ROWS, D_, D_, 1.0f, 0, 0,0, 0,0, 0,0, 1, 1, stream);

        vit_layernorm<<<dim3(ROWS), dim3(256), 0, stream>>>(xbuf, ln2_g + i * D_, ln2_b + i * D_, hbuf, D_);

        // m = gelu(h @ w1 + b1);  x = x + m @ w2 + b2
        launch_gemm(hbuf, D_, W1, MLP_, 0, b1 + i * MLP_, nullptr, big, MLP_,
                    ROWS, MLP_, D_, 1.0f, /*gelu=*/1, 0,0, 0,0, 0,0, 1, 1, stream);
        launch_gemm(big, MLP_, W2, D_, 0, b2 + i * D_, xbuf, xbuf, D_,
                    ROWS, D_, MLP_, 1.0f, 0, 0,0, 0,0, 0,0, 1, 1, stream);
    }

    // 3) head: mean over tokens -> LN -> linear
    vit_meanpool<<<dim3((B_ * D_ + 255) / 256), dim3(256), 0, stream>>>(xbuf, pool);
    vit_layernorm<<<dim3(B_), dim3(256), 0, stream>>>(pool, hn_g, hn_b, pool2, D_);
    launch_gemm(pool2, D_, hw, NC_, 0, hb, nullptr, out, NC_,
                B_, NC_, D_, 1.0f, 0, 0,0, 0,0, 0,0, 1, 1, stream);
}
